// MultiHeadedSelfAttentionModule_70703751627041
// MI455X (gfx1250) — compile-verified, matrix operand order checked by reading, not run on hardware
//
#include <hip/hip_runtime.h>
#include <hip/hip_bf16.h>

typedef _Float16 half_t;
typedef __attribute__((ext_vector_type(16))) _Float16 v16h;
typedef __attribute__((ext_vector_type(8)))  _Float16 v8h;
typedef __attribute__((ext_vector_type(8)))  float    v8f;
typedef __attribute__((ext_vector_type(4)))  unsigned int u32x4;
typedef __attribute__((ext_vector_type(4)))  int          i32x4;
typedef __attribute__((ext_vector_type(8)))  int          i32x8;

// Problem constants
constexpr int Bc  = 2;
constexpr int Sc  = 2048;
constexpr int Dc  = 1024;
constexpr int Hc  = 16;
constexpr int DHc = 64;
constexpr int NBc = 32;   // S / 64 blocks
constexpr int KKc = 16;   // top-k blocks kept

// Tensor Data Mover availability (probe-verified builtin; arity differs by
// toolchain: ROCm 7.2 / clang-22 = 5 args, amdgpu-toolchain / clang-23 = 6).
#if defined(__has_builtin)
#if __has_builtin(__builtin_amdgcn_tensor_load_to_lds)
#define HAVE_TDM 1
#endif
#endif

// ---------------------------------------------------------------------------
// WMMA helpers
// ---------------------------------------------------------------------------
__device__ __forceinline__ v8f wmma16(v16h a, v16h b, v8f c) {
  // D = A(16x32 f16) * B(32x16 f16) + C(16x16 f32)
  return __builtin_amdgcn_wmma_f32_16x16x32_f16(
      /*neg_a=*/false, a, /*neg_b=*/false, b,
      /*c_mod=*/(short)0, c, /*reuse_a=*/false, /*reuse_b=*/false);
}

// Load a 16-half A/B fragment for one lane from a row pointer with the CDNA5
// 16-bit A-matrix layout: element e -> k = (e&7) + (e>=8 ? 16 : 0) + hi*8.
// Both 8-half runs are contiguous -> two 16-byte loads.
__device__ __forceinline__ v16h load_frag(const half_t* rowp, int hi) {
  v8h lo = *(const v8h*)(rowp + hi * 8);
  v8h hh = *(const v8h*)(rowp + 16 + hi * 8);
  v16h f;
#pragma unroll
  for (int i = 0; i < 8; ++i) { f[i] = lo[i]; f[i + 8] = hh[i]; }
  return f;
}

// ---------------------------------------------------------------------------
// 1) LayerNorm (fp32 in) -> f16 x
// ---------------------------------------------------------------------------
__global__ __launch_bounds__(256) void ln_kernel(const float* __restrict__ x,
                                                 const float* __restrict__ g,
                                                 const float* __restrict__ bb,
                                                 half_t* __restrict__ xh) {
  int row = blockIdx.x;          // B*S rows
  int tid = threadIdx.x;
  __shared__ float r1[256], r2[256];
  const float* xr = x + (size_t)row * Dc;
  float s = 0.f, s2 = 0.f;
  for (int i = tid; i < Dc; i += 256) { float v = xr[i]; s += v; s2 += v * v; }
  r1[tid] = s; r2[tid] = s2;
  __syncthreads();
  for (int off = 128; off > 0; off >>= 1) {
    if (tid < off) { r1[tid] += r1[tid + off]; r2[tid] += r2[tid + off]; }
    __syncthreads();
  }
  float mu = r1[0] * (1.0f / Dc);
  float var = r2[0] * (1.0f / Dc) - mu * mu;
  float rs = rsqrtf(var + 1e-5f);
  for (int i = tid; i < Dc; i += 256)
    xh[(size_t)row * Dc + i] = (half_t)((xr[i] - mu) * rs * g[i] + bb[i]);
}

// ---------------------------------------------------------------------------
// 2) fp32 -> f16 weight conversion
// ---------------------------------------------------------------------------
__global__ __launch_bounds__(256) void cvt_kernel(const float* __restrict__ s,
                                                  half_t* __restrict__ d, int n) {
  int i = blockIdx.x * 256 + threadIdx.x;
  if (i < n) d[i] = (half_t)s[i];
}

// ---------------------------------------------------------------------------
// 3) WMMA GEMM: C[M,N] = A[M,K](f16) * W[K,N](f16) + bias
//    mode 0: scatter f16 to [B,H,S,DH]  (QKV)
//    mode 1: fp32 row-major out (final projection)
//    64x64 macro tile per 256-thread block (8 waves x two 16x16 tiles)
// ---------------------------------------------------------------------------
__global__ __launch_bounds__(256) void gemm64_kernel(
    const half_t* __restrict__ A, const half_t* __restrict__ W,
    const float* __restrict__ bias, half_t* __restrict__ dstH,
    float* __restrict__ dstF, int M, int N, int K, int mode) {
  constexpr int LS = 40;  // half stride: 80B rows -> 16B aligned vector slots
  __shared__ alignas(16) half_t As[64 * LS];
  __shared__ alignas(16) half_t Bs[64 * LS];  // N-major (transposed W tile)

  int tid = threadIdx.x;
  int w = tid >> 5, lane = tid & 31, l16 = lane & 15, hi = lane >> 4;
  int n0 = blockIdx.x * 64, m0 = blockIdx.y * 64;
  int tm = w & 3, tn0 = w >> 2;

  v8f acc0 = {0.f, 0.f, 0.f, 0.f, 0.f, 0.f, 0.f, 0.f};
  v8f acc1 = {0.f, 0.f, 0.f, 0.f, 0.f, 0.f, 0.f, 0.f};

  for (int kt = 0; kt < K; kt += 32) {
    {  // stage A 64x32 (8 contiguous halves / thread)
      int r = tid >> 2, c = (tid & 3) * 8;
      const half_t* src = A + (size_t)(m0 + r) * K + kt + c;
      *(v8h*)(As + r * LS + c) = *(const v8h*)src;
      if (kt + 32 < K) __builtin_prefetch(src + 32, 0, 1);  // global_prefetch
    }
    {  // stage W 32x64 transposed -> Bs[n][k]
      int k = tid >> 3, nc = (tid & 7) * 8;
      const half_t* src = W + (size_t)(kt + k) * N + n0 + nc;
      v8h wv = *(const v8h*)src;
      if (kt + 32 < K) __builtin_prefetch(W + (size_t)(kt + 32 + k) * N + n0 + nc, 0, 1);
#pragma unroll
      for (int i = 0; i < 8; ++i) Bs[(nc + i) * LS + k] = wv[i];
    }
    __syncthreads();
    v16h a  = load_frag(As + (tm * 16 + l16) * LS, hi);
    v16h b0 = load_frag(Bs + (tn0 * 16 + l16) * LS, hi);
    v16h b1 = load_frag(Bs + ((tn0 + 2) * 16 + l16) * LS, hi);
    acc0 = wmma16(a, b0, acc0);
    acc1 = wmma16(a, b1, acc1);
    __syncthreads();
  }

#pragma unroll
  for (int t = 0; t < 2; ++t) {
    v8f acc = t ? acc1 : acc0;
    int tnn = tn0 + t * 2;
#pragma unroll
    for (int r = 0; r < 8; ++r) {
      int gm = m0 + tm * 16 + r + 8 * hi;
      int gn = n0 + tnn * 16 + l16;
      float val = acc[r] + bias[gn];
      if (mode == 0) {
        int bb = gm >> 11, ss = gm & (Sc - 1);   // S = 2048
        int hh = gn >> 6, dd = gn & (DHc - 1);   // DH = 64
        dstH[(((size_t)bb * Hc + hh) * Sc + ss) * DHc + dd] = (half_t)val;
      } else {
        dstF[(size_t)gm * N + gn] = val;
      }
    }
  }
}

// ---------------------------------------------------------------------------
// 4) Per-block means of Q and K: [B,H,NB,DH] fp32
// ---------------------------------------------------------------------------
__global__ __launch_bounds__(64) void bmean_kernel(const half_t* __restrict__ qh,
                                                   const half_t* __restrict__ kh,
                                                   float* __restrict__ qm,
                                                   float* __restrict__ km) {
  int blk = blockIdx.x;              // bh*NB + nb
  int d = threadIdx.x;               // 0..63
  int nb = blk % NBc, bh = blk / NBc;
  size_t base = ((size_t)bh * Sc + (size_t)nb * 64) * DHc;
  float sq = 0.f, sk = 0.f;
#pragma unroll 4
  for (int r = 0; r < 64; ++r) {
    sq += (float)qh[base + (size_t)r * DHc + d];
    sk += (float)kh[base + (size_t)r * DHc + d];
  }
  qm[(size_t)blk * DHc + d] = sq * (1.0f / 64.0f);
  km[(size_t)blk * DHc + d] = sk * (1.0f / 64.0f);
}

// ---------------------------------------------------------------------------
// 5) Top-k block mask: one wave32 per (b,h,qb); nb=32 fits one uint32 mask.
// ---------------------------------------------------------------------------
__global__ __launch_bounds__(32) void topk_kernel(const float* __restrict__ qm,
                                                  const float* __restrict__ km,
                                                  unsigned* __restrict__ mask) {
  int blk = blockIdx.x;              // bh*NB + qb
  int qb = blk % NBc, bh = blk / NBc;
  int j = threadIdx.x;               // candidate KV block = lane
  const float* qv = qm + ((size_t)bh * NBc + qb) * DHc;
  const float* kv = km + ((size_t)bh * NBc + j) * DHc;
  float acc = 0.f;
#pragma unroll 8
  for (int d = 0; d < DHc; ++d) acc += qv[d] * kv[d];
  float sim = (j <= qb) ? acc : -1e9f;
  int rank = 0;
  for (int i = 0; i < 32; ++i) {
    float si = __shfl(sim, i, 32);
    if (si > sim || (si == sim && i < j)) rank++;
  }
  bool sel = (j <= qb) && (rank < KKc);
  unsigned mw = (unsigned)(__ballot(sel) & 0xffffffffull);
  mw |= (1u << qb);                  // always keep diagonal block
  if (j == 0) mask[blk] = mw;
}

// ---------------------------------------------------------------------------
// 6) Block-sparse flash attention. One 128-thread block per (b,h,qb).
//    Each wave owns 16 query rows; scores & P*V via WMMA, online softmax.
//    K tile staged by the Tensor Data Mover (2D descriptor, LDS row padding
//    32 DW + 4 DW pad -> 72-half stride); V staged+transposed cooperatively.
// ---------------------------------------------------------------------------
__global__ __launch_bounds__(128) void attn_kernel(
    const half_t* __restrict__ qh, const half_t* __restrict__ kh,
    const half_t* __restrict__ vh, const unsigned* __restrict__ maskbuf,
    half_t* __restrict__ ctxh) {
  constexpr int LST = 72;  // 144B row stride, 16B aligned
  __shared__ alignas(16) half_t Ks[64 * LST];  // K[kv][dh]
  __shared__ alignas(16) half_t Vs[64 * LST];  // V^T[dh][kv]
  __shared__ alignas(16) half_t Ps[64 * LST];  // P[q][kv] layout round-trip

  int blk = blockIdx.x;
  int qb = blk % NBc, bh = blk / NBc;
  int b = bh / Hc, h = bh % Hc;
  int tid = threadIdx.x, w = tid >> 5, lane = tid & 31;
  int l16 = lane & 15, hi = lane >> 4;

  size_t headbase = (size_t)bh * Sc * DHc;

  // Q fragments for this wave's 16 rows (row = qb*64 + w*16 + l16)
  const half_t* qrow = qh + headbase + (size_t)(qb * 64 + w * 16 + l16) * DHc;
  v16h af0 = load_frag(qrow + 0, hi);
  v16h af1 = load_frag(qrow + 32, hi);

  v8f od[4] = {};            // output accumulators over dh tiles
  float mrow[8], lrow[8];
#pragma unroll
  for (int r = 0; r < 8; ++r) { mrow[r] = -1e30f; lrow[r] = 0.f; }

  unsigned mb = maskbuf[blk];

  for (int j = 0; j < NBc; ++j) {
    if (!((mb >> j) & 1u)) continue;  // uniform across block

#ifdef HAVE_TDM
    if (w == 0) {
      // TDM: 64x64 f16 tile of K, row stride 64 halves in global, LDS rows
      // padded to 72 halves (pad_interval=32 DW -> code 4, pad_amount=4 DW
      // -> code 3). One descriptor, one DMA, tracked by TENSORcnt.
      unsigned lds_base = (unsigned)(uintptr_t)(void*)Ks;
      unsigned long long ga =
          (unsigned long long)(uintptr_t)(kh + headbase + (size_t)j * 64 * DHc);
      u32x4 g0;
      g0[0] = 1u;                                         // count=1, user mode
      g0[1] = lds_base;                                   // lds_addr
      g0[2] = (unsigned)(ga & 0xffffffffu);               // global_addr[31:0]
      g0[3] = (unsigned)((ga >> 32) & 0x01ffffffu)        // global_addr[56:32]
              | (2u << 30);                               // type=2 (image)
      i32x8 g1;
      g1[0] = (int)((1u << 16) |                          // data_size=2B
                    (1u << 20) |                          // pad_enable
                    (4u << 22) |                          // pad_interval=32 DW
                    (3u << 25));                          // pad_amount=4 DW
      g1[1] = (int)(64u << 16);                           // tensor_dim0=64
      g1[2] = (int)((2048u & 0xffffu) << 16);             // tensor_dim1=2048
      g1[3] = (int)(64u << 16);                           // tile_dim0=64
      g1[4] = 64;                                         // tile_dim1=64
      g1[5] = 64;                                         // dim0_stride=64
      g1[6] = 0;
      g1[7] = 0;
      i32x4 gz = {0, 0, 0, 0};                            // groups 2/3 unused
#if __clang_major__ >= 23
      i32x8 gz8 = {0, 0, 0, 0, 0, 0, 0, 0};
      __builtin_amdgcn_tensor_load_to_lds(g0, g1, gz, gz, gz8, 0);
#else
      __builtin_amdgcn_tensor_load_to_lds(g0, g1, gz, gz, 0);
#endif
#if __has_builtin(__builtin_amdgcn_s_wait_tensorcnt)
      __builtin_amdgcn_s_wait_tensorcnt(0);
#else
      asm volatile("s_wait_tensorcnt 0" ::: "memory");
#endif
    }
    {  // stage V^T tile cooperatively (TDM cannot transpose)
      int r = tid >> 1, off = (tid & 1) * 32;
      const half_t* vrow = vh + headbase + (size_t)(j * 64 + r) * DHc + off;
#pragma unroll
      for (int i = 0; i < 4; ++i) {
        v8h vv = *(const v8h*)(vrow + i * 8);
#pragma unroll
        for (int e = 0; e < 8; ++e) Vs[(off + i * 8 + e) * LST + r] = vv[e];
      }
    }
#else
    {  // fallback: stage K tile and V^T tile cooperatively
      int r = tid >> 1, off = (tid & 1) * 32;
      const half_t* krow = kh + headbase + (size_t)(j * 64 + r) * DHc + off;
      const half_t* vrow = vh + headbase + (size_t)(j * 64 + r) * DHc + off;
#pragma unroll
      for (int i = 0; i < 4; ++i)
        *(v8h*)(Ks + r * LST + off + i * 8) = *(const v8h*)(krow + i * 8);
#pragma unroll
      for (int i = 0; i < 4; ++i) {
        v8h vv = *(const v8h*)(vrow + i * 8);
#pragma unroll
        for (int e = 0; e < 8; ++e) Vs[(off + i * 8 + e) * LST + r] = vv[e];
      }
    }
#endif
    __syncthreads();

    // scores: S = Q * K^T  (4 N-tiles of 16, K-dim = 64 -> 2 WMMA each)
    v8f sc[4];
#pragma unroll
    for (int tn = 0; tn < 4; ++tn) {
      v16h bk0 = load_frag(Ks + (tn * 16 + l16) * LST + 0, hi);
      v16h bk1 = load_frag(Ks + (tn * 16 + l16) * LST + 32, hi);
      v8f z = {0.f, 0.f, 0.f, 0.f, 0.f, 0.f, 0.f, 0.f};
      z = wmma16(af0, bk0, z);
      z = wmma16(af1, bk1, z);
      sc[tn] = z;
    }

    // scale + causal mask (only needed on diagonal block) + row max
    float mnew[8];
#pragma unroll
    for (int r = 0; r < 8; ++r) {
      float cand = -1e30f;
#pragma unroll
      for (int tn = 0; tn < 4; ++tn) {
        float s = sc[tn][r] * 0.125f;  // 1/sqrt(64)
        if (j == qb && (tn * 16 + l16) > (w * 16 + r + 8 * hi)) s = -1e9f;
        sc[tn][r] = s;
        cand = fmaxf(cand, s);
      }
#pragma unroll
      for (int off = 1; off < 16; off <<= 1)
        cand = fmaxf(cand, __shfl_xor(cand, off, 32));  // 16-lane group reduce
      mnew[r] = fmaxf(mrow[r], cand);
    }

    // rescale old state, exponentiate, accumulate row sums, spill P to LDS
#pragma unroll
    for (int r = 0; r < 8; ++r) {
      float alpha = __expf(mrow[r] - mnew[r]);
      mrow[r] = mnew[r];
      lrow[r] *= alpha;
#pragma unroll
      for (int tn = 0; tn < 4; ++tn) od[tn][r] *= alpha;
      float rs = 0.f;
#pragma unroll
      for (int tn = 0; tn < 4; ++tn) {
        float p = __expf(sc[tn][r] - mrow[r]);
        rs += p;
        Ps[(w * 16 + r + 8 * hi) * LST + tn * 16 + l16] = (half_t)p;
      }
#pragma unroll
      for (int off = 1; off < 16; off <<= 1) rs += __shfl_xor(rs, off, 32);
      lrow[r] += rs;
    }

    // intra-wave LDS RAW: P spilled above is reloaded in A-fragment layout
    asm volatile("s_wait_dscnt 0" ::: "memory");

    v16h pf0 = load_frag(Ps + (w * 16 + l16) * LST + 0, hi);
    v16h pf1 = load_frag(Ps + (w * 16 + l16) * LST + 32, hi);

    // O += P * V   (4 dh tiles, K-dim = 64 kv -> 2 WMMA each)
#pragma unroll
    for (int tn = 0; tn < 4; ++tn) {
      v16h bv0 = load_frag(Vs + (tn * 16 + l16) * LST + 0, hi);
      v16h bv1 = load_frag(Vs + (tn * 16 + l16) * LST + 32, hi);
      od[tn] = wmma16(pf0, bv0, od[tn]);
      od[tn] = wmma16(pf1, bv1, od[tn]);
    }
    __syncthreads();
  }

  // epilogue: normalize and scatter ctx to [B,S,D] f16 (D = H*DH)
#pragma unroll
  for (int r = 0; r < 8; ++r) {
    float inv = 1.0f / fmaxf(lrow[r], 1e-20f);
    int srow = qb * 64 + w * 16 + r + 8 * hi;
#pragma unroll
    for (int tn = 0; tn < 4; ++tn) {
      int col = h * DHc + tn * 16 + l16;
      ctxh[((size_t)b * Sc + srow) * Dc + col] = (half_t)(od[tn][r] * inv);
    }
  }
}

// ---------------------------------------------------------------------------
// Launch
// ---------------------------------------------------------------------------
extern "C" void kernel_launch(void* const* d_in, const int* in_sizes, int n_in,
                              void* d_out, int out_size, void* d_ws, size_t ws_size,
                              hipStream_t stream) {
  const float* x    = (const float*)d_in[0];
  const float* ln_g = (const float*)d_in[1];
  const float* ln_b = (const float*)d_in[2];
  const float* Wq   = (const float*)d_in[3];
  const float* bq   = (const float*)d_in[4];
  const float* Wk   = (const float*)d_in[5];
  const float* bk   = (const float*)d_in[6];
  const float* Wv   = (const float*)d_in[7];
  const float* bv   = (const float*)d_in[8];
  const float* Wo   = (const float*)d_in[9];
  const float* bo   = (const float*)d_in[10];
  float* out = (float*)d_out;

  char* ws = (char*)d_ws;
  const size_t xh_b  = (size_t)Bc * Sc * Dc * 2;   // 8 MB
  const size_t w_b   = (size_t)Dc * Dc * 2;        // 2 MB
  const size_t q_b   = (size_t)Bc * Hc * Sc * DHc * 2;  // 8 MB
  const size_t m_b   = (size_t)Bc * Hc * NBc * DHc * 4; // 256 KB

  size_t off = 0;
  half_t* xh  = (half_t*)(ws + off); off += xh_b;
  half_t* wqh = (half_t*)(ws + off); off += w_b;
  half_t* wkh = (half_t*)(ws + off); off += w_b;
  half_t* wvh = (half_t*)(ws + off); off += w_b;
  half_t* woh = (half_t*)(ws + off); off += w_b;
  half_t* qhp = (half_t*)(ws + off); off += q_b;
  half_t* khp = (half_t*)(ws + off); off += q_b;
  half_t* vhp = (half_t*)(ws + off); off += q_b;
  half_t* ctx = (half_t*)(ws + off); off += xh_b;
  float*  qm  = (float*)(ws + off);  off += m_b;
  float*  km  = (float*)(ws + off);  off += m_b;
  unsigned* mask = (unsigned*)(ws + off); off += (size_t)Bc * Hc * NBc * 4;

  // 1) LayerNorm + cast
  ln_kernel<<<Bc * Sc, 256, 0, stream>>>(x, ln_g, ln_b, xh);

  // 2) Weight conversion
  int nW = Dc * Dc;
  cvt_kernel<<<(nW + 255) / 256, 256, 0, stream>>>(Wq, wqh, nW);
  cvt_kernel<<<(nW + 255) / 256, 256, 0, stream>>>(Wk, wkh, nW);
  cvt_kernel<<<(nW + 255) / 256, 256, 0, stream>>>(Wv, wvh, nW);
  cvt_kernel<<<(nW + 255) / 256, 256, 0, stream>>>(Wo, woh, nW);

  // 3) QKV projections (WMMA GEMM, scatter to [B,H,S,DH] f16)
  dim3 gg(Dc / 64, (Bc * Sc) / 64);
  gemm64_kernel<<<gg, 256, 0, stream>>>(xh, wqh, bq, qhp, nullptr,
                                        Bc * Sc, Dc, Dc, 0);
  gemm64_kernel<<<gg, 256, 0, stream>>>(xh, wkh, bk, khp, nullptr,
                                        Bc * Sc, Dc, Dc, 0);
  gemm64_kernel<<<gg, 256, 0, stream>>>(xh, wvh, bv, vhp, nullptr,
                                        Bc * Sc, Dc, Dc, 0);

  // 4) Block means, 5) top-k block bitmask
  bmean_kernel<<<Bc * Hc * NBc, 64, 0, stream>>>(qhp, khp, qm, km);
  topk_kernel<<<Bc * Hc * NBc, 32, 0, stream>>>(qm, km, mask);

  // 6) Block-sparse flash attention (WMMA + TDM K staging)
  attn_kernel<<<Bc * Hc * NBc, 128, 0, stream>>>(qhp, khp, vhp, mask, ctx);

  // 7) Output projection (WMMA GEMM, fp32 out + bias)
  gemm64_kernel<<<gg, 256, 0, stream>>>(ctx, woh, bo, nullptr, out,
                                        Bc * Sc, Dc, Dc, 1);

  (void)in_sizes; (void)n_in; (void)out_size; (void)ws_size;
}
